// VectorQuantizer_3109556323066
// MI455X (gfx1250) — matrix operand order repaired and checked by reading, not compiled, not run on hardware
//
#include <hip/hip_runtime.h>
#include <hip/hip_bf16.h>

typedef __attribute__((ext_vector_type(8)))  _Float16 v8h;
typedef __attribute__((ext_vector_type(16))) _Float16 v16h;
typedef __attribute__((ext_vector_type(8)))  float    v8f;
typedef unsigned int v4u __attribute__((ext_vector_type(4)));
typedef int          v4i __attribute__((ext_vector_type(4)));
typedef int          v8i __attribute__((ext_vector_type(8)));

// Problem constants (from reference)
constexpr int D        = 64;        // embedding dim
constexpr int K        = 1024;      // codebook size
constexpr int HW       = 4096;      // H*W
constexpr int NTOT     = 131072;    // B*H*W flattened rows
constexpr int QELEMS   = NTOT * D;  // 8388608 floats of quantized output
constexpr int LOSS_OFF = QELEMS;    // scalar loss slot
constexpr int IDX_OFF  = QELEMS + 1;// indices (stored as float)

// Tiling
constexpr int WAVES         = 4;          // 128 threads
constexpr int MT            = 4;          // 16-row M tiles per wave
constexpr int ROWS_PER_WAVE = MT * 16;    // 64
constexpr int ROWS_PER_WG   = WAVES * ROWS_PER_WAVE; // 256
constexpr int EPAD_X        = 72;         // padded f16 stride of an x row in LDS
constexpr int CHUNK_CODES   = 128;        // codebook codes per TDM chunk (32 KB f32)
constexpr int NCHUNK        = K / CHUNK_CODES; // 8

__device__ __forceinline__ v16h pack16(const _Float16* plo, const _Float16* phi) {
  v8h lo = *reinterpret_cast<const v8h*>(plo);
  v8h hi = *reinterpret_cast<const v8h*>(phi);
  v16h r;
#pragma unroll
  for (int i = 0; i < 8; ++i) { r[i] = lo[i]; r[i + 8] = hi[i]; }
  return r;
}

// 1-D Tensor Data Mover copy: n_qwords * 8 bytes from global -> LDS.
// D# layout per CDNA5 ISA (group0: count/lds/global/type,
// group1: data_size=8B, tensor_dim0 = tile_dim0 = n_qwords, 1-D).
__device__ __forceinline__ void tdm_load_1d(const void* gsrc, unsigned lds_byte_off,
                                            unsigned n_qwords) {
  unsigned long long ga = (unsigned long long)(uintptr_t)gsrc;
  v4u g0;
  g0.x = 1u;                                               // count=1 (user)
  g0.y = lds_byte_off;                                     // lds_addr (bytes)
  g0.z = (unsigned)(ga & 0xffffffffu);                     // global_addr[31:0]
  g0.w = (unsigned)((ga >> 32) & 0x01ffffffu) | (2u << 30);// addr[56:32] | type=2
  v8i g1;
  g1[0] = (3 << 16);                                       // data_size = 8B
  g1[1] = (int)((n_qwords & 0xffffu) << 16);               // tensor_dim0[15:0]
  g1[2] = (int)((n_qwords >> 16) & 0xffffu) | (1 << 16);   // dim0[31:16], dim1=1
  g1[3] = (int)((n_qwords & 0xffffu) << 16);               // tile_dim0
  g1[4] = 0;                                               // tile_dim1/2 unused
  g1[5] = (int)n_qwords;                                   // dim0_stride lo
  g1[6] = 0;
  g1[7] = 0;
  v4i gz = {0, 0, 0, 0};
#if defined(__clang_major__) && __clang_major__ >= 23
  v8i gz8 = {0, 0, 0, 0, 0, 0, 0, 0};
  __builtin_amdgcn_tensor_load_to_lds(g0, g1, gz, gz, gz8, 0);
#else
  __builtin_amdgcn_tensor_load_to_lds(g0, g1, gz, gz, 0);
#endif
}

__global__ __launch_bounds__(128)
void vq_wmma_kernel(const float* __restrict__ in, const float* __restrict__ eg,
                    float* __restrict__ out)
{
  __shared__ __align__(16) _Float16 s_emb[K * D];                          // 128 KB f16 codebook
  __shared__ __align__(16) float    s_stage[2][CHUNK_CODES * D];           //  64 KB TDM double buffer
  __shared__ __align__(16) _Float16 s_x[WAVES * ROWS_PER_WAVE * EPAD_X];   //  36 KB x tiles
  __shared__ float                  s_enorm[K];                            //   4 KB
  __shared__ int                    s_idx[WAVES * ROWS_PER_WAVE];          //   1 KB

  const int tid  = threadIdx.x;
  const int lane = tid & 31;
  const int wave = tid >> 5;
  const int gh   = lane >> 4;   // half-wave group (0: lanes 0-15, 1: lanes 16-31)
  const int l15  = lane & 15;

  constexpr unsigned CHUNK_QW = (unsigned)(CHUNK_CODES * D / 2); // 4096 qwords = 32 KB

  // ---- Kick TDM on the first codebook chunk (wave 0 drives the DMA) ----
  if (wave == 0) {
    tdm_load_1d(eg, (unsigned)(uintptr_t)&s_stage[0][0], CHUNK_QW);
  }

  // ---- Overlap: stage this wave's 64 input rows, transposed, f32 -> f16 ----
  // inputs layout [B, D, H, W]: element (n, d) at b*D*HW + d*HW + (n % HW)
  const int n0  = (blockIdx.x * WAVES + wave) * ROWS_PER_WAVE;
  const int b   = n0 >> 12;          // n0 / HW
  const int hw0 = n0 & (HW - 1);
  const float* xin = in + (size_t)b * (D * HW) + hw0;
  _Float16* sx = s_x + wave * ROWS_PER_WAVE * EPAD_X;
#pragma unroll 4
  for (int d = 0; d < D; ++d) {
    sx[lane * EPAD_X + d]        = (_Float16)xin[(size_t)d * HW + lane];
    sx[(lane + 32) * EPAD_X + d] = (_Float16)xin[(size_t)d * HW + lane + 32];
  }

  // ---- Pipelined TDM chunks -> f16 conversion into s_emb ----
  for (int ch = 0; ch < NCHUNK; ++ch) {
    if (wave == 0) {
      if (ch + 1 < NCHUNK) {
        tdm_load_1d(eg + (size_t)(ch + 1) * (CHUNK_CODES * D),
                    (unsigned)(uintptr_t)&s_stage[(ch + 1) & 1][0], CHUNK_QW);
        __builtin_amdgcn_s_wait_tensorcnt((short)1); // chunk ch complete
      } else {
        __builtin_amdgcn_s_wait_tensorcnt((short)0);
      }
    }
    __syncthreads();
    const float2* st2 = reinterpret_cast<const float2*>(&s_stage[ch & 1][0]);
    _Float16* dst = s_emb + ch * (CHUNK_CODES * D);
    for (int i = tid; i < (CHUNK_CODES * D) / 2; i += 128) {
      float2 v = st2[i];
      dst[2 * i]     = (_Float16)v.x;
      dst[2 * i + 1] = (_Float16)v.y;
    }
    __syncthreads();
  }

  // ---- Code norms from the f16 values (consistent with f16 dot products) ----
  for (int c = tid; c < K; c += 128) {
    const _Float16* p = s_emb + c * D;
    float s = 0.f;
#pragma unroll
    for (int d = 0; d < D; ++d) { float v = (float)p[d]; s += v * v; }
    s_enorm[c] = s;
  }
  __syncthreads();

  // ---- Build A fragments (MT M-tiles x 2 K-halves), WMMA 16-bit A layout:
  // lanes 0-15 row m hold K {0..7, 16..23}; lanes 16-31 hold K {8..15, 24..31}
  v16h a[MT][2];
#pragma unroll
  for (int t = 0; t < MT; ++t) {
    const _Float16* pr = sx + (t * 16 + l15) * EPAD_X;
#pragma unroll
    for (int kh = 0; kh < 2; ++kh) {
      const _Float16* p0 = pr + kh * 32 + gh * 8;
      a[t][kh] = pack16(p0, p0 + 16);
    }
  }

  // ---- Distance scan: argmin_c ( ||e_c||^2 - 2 x.e_c )  (||x||^2 is row-const)
  float best[MT][8];
  int   bidx[MT][8];
#pragma unroll
  for (int t = 0; t < MT; ++t)
#pragma unroll
    for (int i = 0; i < 8; ++i) { best[t][i] = 3.4e38f; bidx[t][i] = 0; }

  for (int ct = 0; ct < K / 16; ++ct) {
    const int c = ct * 16 + l15;
    // B layout 32x16: lanes 0-15 col=lane K {0..7, 16..23}; lanes 16-31 col=lane-16 K {8..15, 24..31}
    const _Float16* bp = s_emb + c * D + gh * 8;
    v16h b0 = pack16(bp, bp + 16);       // d = 0..31
    v16h b1 = pack16(bp + 32, bp + 48);  // d = 32..63
    float en = s_enorm[c];
#pragma unroll
    for (int t = 0; t < MT; ++t) {
      v8f acc = {};
      acc = __builtin_amdgcn_wmma_f32_16x16x32_f16(false, a[t][0], false, b0,
                                                   (short)0, acc, false, false);
      acc = __builtin_amdgcn_wmma_f32_16x16x32_f16(false, a[t][1], false, b1,
                                                   (short)0, acc, false, false);
#pragma unroll
      for (int i = 0; i < 8; ++i) {
        float dist = en - 2.f * acc[i];   // row M = t*16 + gh*8 + i, col = c
        if (dist < best[t][i]) { best[t][i] = dist; bidx[t][i] = c; }
      }
    }
  }

  // ---- Reduce argmin across the 16 lanes of each half-wave (xor butterfly) ----
#pragma unroll
  for (int mask = 8; mask >= 1; mask >>= 1) {
#pragma unroll
    for (int t = 0; t < MT; ++t)
#pragma unroll
      for (int i = 0; i < 8; ++i) {
        float ov = __shfl_xor(best[t][i], mask, 32);
        int   oi = __shfl_xor(bidx[t][i], mask, 32);
        if (ov < best[t][i] || (ov == best[t][i] && oi < bidx[t][i])) {
          best[t][i] = ov; bidx[t][i] = oi;
        }
      }
  }

  // lanes 0-15 hold rows {t*16+0..7}; lanes 16-31 hold rows {t*16+8..15}
  if (l15 == 0) {
    const int rofs = gh * 8;
#pragma unroll
    for (int t = 0; t < MT; ++t)
#pragma unroll
      for (int i = 0; i < 8; ++i) {
        int row = t * 16 + rofs + i;
        s_idx[wave * ROWS_PER_WAVE + row] = bidx[t][i];
        out[IDX_OFF + n0 + row] = (float)bidx[t][i];
      }
  }
  __syncthreads();

  // ---- Gather quantized rows (exact f32 from global/L2), write + loss ----
  const int r0 = gh * 16 + l15;      // rows 0..31 across lanes
  const int r1 = r0 + 32;            // rows 32..63
  const int c0 = s_idx[wave * ROWS_PER_WAVE + r0];
  const int c1 = s_idx[wave * ROWS_PER_WAVE + r1];
  const float* e0 = eg + (size_t)c0 * D;
  const float* e1 = eg + (size_t)c1 * D;
  float* qout = out + (size_t)b * (D * HW) + hw0;
  float lsum = 0.f;
#pragma unroll 4
  for (int d = 0; d < D; ++d) {
    size_t base = (size_t)d * HW;
    float ev0 = e0[d], ev1 = e1[d];
    float x0 = xin[base + r0], x1 = xin[base + r1];
    qout[base + r0] = ev0;           // coalesced over 32 consecutive rows
    qout[base + r1] = ev1;
    float d0 = ev0 - x0, d1 = ev1 - x1;
    lsum += d0 * d0 + d1 * d1;
  }
  lsum += __shfl_xor(lsum, 16, 32);
  lsum += __shfl_xor(lsum, 8, 32);
  lsum += __shfl_xor(lsum, 4, 32);
  lsum += __shfl_xor(lsum, 2, 32);
  lsum += __shfl_xor(lsum, 1, 32);
  if (lane == 0) {
    __hip_atomic_fetch_add(out + LOSS_OFF, lsum * (0.25f / (float)QELEMS),
                           __ATOMIC_RELAXED, __HIP_MEMORY_SCOPE_AGENT);
  }
}

extern "C" void kernel_launch(void* const* d_in, const int* in_sizes, int n_in,
                              void* d_out, int out_size, void* d_ws, size_t ws_size,
                              hipStream_t stream) {
  const float* inp = (const float*)d_in[0];   // [32, 64, 64, 64] f32
  const float* emb = (const float*)d_in[1];   // [1024, 64] f32
  float* out = (float*)d_out;                 // [quantized | loss | indices(as f32)]

  // zero the loss accumulator (capture-safe)
  hipMemsetAsync(out + LOSS_OFF, 0, sizeof(float), stream);

  vq_wmma_kernel<<<dim3(NTOT / ROWS_PER_WG), dim3(128), 0, stream>>>(inp, emb, out);
}
